// LocalPatternBinaryClassifier_84894323573140
// MI455X (gfx1250) — matrix operand order repaired and checked by reading, not compiled, not run on hardware
//
#include <hip/hip_runtime.h>
#include <hip/hip_bf16.h>
#include <math.h>

typedef __attribute__((ext_vector_type(16))) _Float16 v16h;
typedef __attribute__((ext_vector_type(8)))  float    v8f;

namespace {
constexpr int Bg  = 16;           // graphs
constexpr int NPG = 1024;         // nodes per graph
constexpr int Nn  = Bg * NPG;     // 16384 nodes
constexpr int Kn  = 12;           // knn
constexpr int DM  = 128;
constexpr int DH  = 256;
constexpr int DI  = 2 * DM + 4;   // 260
constexpr int KP  = 288;          // 260 padded to 9 k-steps of 32
constexpr int En  = Nn * Kn;      // 196608 edges
// packed f16 weight region (halves), per layer: msgW1 | msgW2 | updW1 | updW2
constexpr int T_MSG1 = (KP / 32) * (DH / 16);   // 144 tiles
constexpr int T_MSG2 = (DH / 32) * (DM / 16);   // 64
constexpr int T_UPD1 = (DH / 32) * (DH / 16);   // 128
constexpr int T_UPD2 = (DH / 32) * (DM / 16);   // 64
constexpr int OFF_MSG2 = T_MSG1 * 512;
constexpr int OFF_UPD1 = OFF_MSG2 + T_MSG2 * 512;
constexpr int OFF_UPD2 = OFF_UPD1 + T_UPD1 * 512;
constexpr int LSTRIDE  = OFF_UPD2 + T_UPD2 * 512;      // 204800 halves / layer
}

__device__ __forceinline__ float silu_f(float v) { return v / (1.f + __expf(-v)); }

// LDS A-fragment load. The offset is laundered through an empty asm so the
// address is loop-variant to the optimizer: LICM cannot hoist the loads out of
// the col-tile loops (which caused scratch spills), yet the base remains the
// __shared__ array so addrspace(3) inference still yields ds_load_b128 (not
// flat/SYS loads as the previous volatile version did).
__device__ __forceinline__ v16h lds_frag(const _Float16* base, int off) {
  asm volatile("" : "+v"(off));
  return *(const v16h*)(base + off);
}

#define WMMA_F16(a, b, c) \
  __builtin_amdgcn_wmma_f32_16x16x32_f16(false, (a), false, (b), (short)0, (c), false, false)

// ---------------- weight pack: f32 (krows x ncols) -> f16 B-fragment tiles ----------------
__global__ void pack_b_kernel(const float* __restrict__ W, _Float16* __restrict__ out,
                              int krows, int ncols, int ctiles) {
  const int tile = blockIdx.x;
  const int st = tile / ctiles, ct = tile - st * ctiles;
  const int lane = threadIdx.x;
  const int n = lane & 15, kh = lane >> 4;
  _Float16* dst = out + (size_t)tile * 512 + lane * 16;
#pragma unroll
  for (int j = 0; j < 16; ++j) {
    const int k = st * 32 + kh * 8 + j + ((j >> 3) << 3);
    dst[j] = (_Float16)((k < krows) ? W[(size_t)k * ncols + ct * 16 + n] : 0.f);
  }
}

// ---------------- kNN + edge attributes ----------------
__global__ void knn_kernel(const float* __restrict__ x, int* __restrict__ srcIdx,
                           float* __restrict__ ea) {
  __shared__ float s_px[NPG], s_py[NPG], s_f[NPG];
  const int g = blockIdx.x >> 2;
  const int chunk = blockIdx.x & 3;
  const int tid = threadIdx.x;
  for (int n = tid; n < NPG; n += 256) {
    const float* xp = x + (size_t)(g * NPG + n) * 3;
    s_f[n] = xp[0]; s_px[n] = xp[1]; s_py[n] = xp[2];
  }
  __syncthreads();
  const int nl = chunk * 256 + tid;
  const float px = s_px[nl], py = s_py[nl], f0 = s_f[nl];
  float bd[Kn]; int bi[Kn];
#pragma unroll
  for (int q = 0; q < Kn; ++q) { bd[q] = 3.0e38f; bi[q] = 0; }
  for (int j = 0; j < NPG; ++j) {
    if (j == nl) continue;
    const float dx = s_px[j] - px, dy = s_py[j] - py;
    float d = dx * dx + dy * dy;
    if (d < bd[Kn - 1]) {
      bd[Kn - 1] = d; bi[Kn - 1] = j;
#pragma unroll
      for (int q = Kn - 1; q > 0; --q) {
        if (bd[q] < bd[q - 1]) {
          float td = bd[q]; bd[q] = bd[q - 1]; bd[q - 1] = td;
          int ti = bi[q]; bi[q] = bi[q - 1]; bi[q - 1] = ti;
        }
      }
    }
  }
  const int node = g * NPG + nl;
#pragma unroll
  for (int q = 0; q < Kn; ++q) {
    const int j = bi[q];
    srcIdx[(size_t)node * Kn + q] = g * NPG + j;
    const float dx = s_px[j] - px, dy = s_py[j] - py;
    float* e = ea + ((size_t)node * Kn + q) * 4;
    e[0] = dx; e[1] = dy;
    e[2] = sqrtf(dx * dx + dy * dy + 1e-12f);
    e[3] = s_f[j] - f0;
  }
}

// ---------------- input MLP ----------------
__global__ void input_mlp_kernel(const float* __restrict__ x,
                                 const float* __restrict__ W1, const float* __restrict__ b1,
                                 const float* __restrict__ W2, const float* __restrict__ b2,
                                 float* __restrict__ h) {
  __shared__ float s_t[DM];
  const int node = blockIdx.x, j = threadIdx.x;
  const float x0 = x[(size_t)node * 3], x1 = x[(size_t)node * 3 + 1], x2 = x[(size_t)node * 3 + 2];
  s_t[j] = silu_f(x0 * W1[j] + x1 * W1[DM + j] + x2 * W1[2 * DM + j] + b1[j]);
  __syncthreads();
  float acc = b2[j];
#pragma unroll 8
  for (int k = 0; k < DM; ++k) acc += s_t[k] * W2[(size_t)k * DM + j];
  h[(size_t)node * DM + j] = acc;
}

// Swizzled A-tile layout: k = 8*a + b -> lane = m + 16*(a&1), elem j = b + 8*(a>>1)

// ---------------- message MLP over edges + segment-sum (WMMA) ----------------
// Block = 4 nodes = 48 edges = 3 row tiles; 128 threads = 4 waves.
__global__ void msg_kernel(const float* __restrict__ h, const float* __restrict__ ea,
                           const int* __restrict__ srcIdx,
                           const _Float16* __restrict__ W1p, const float* __restrict__ b1,
                           const _Float16* __restrict__ W2p, const float* __restrict__ b2,
                           float* __restrict__ agg) {
  __shared__ __align__(32) _Float16 s_in[3 * 9 * 512];   // A tiles (rt, st), swizzled
  __shared__ __align__(32) _Float16 s_t[3 * 8 * 512];    // A tiles for stage 2
  float* s_m = (float*)s_in;                             // aliased after stage 1
  const int tid = threadIdx.x;
  const int wave = tid >> 5;
  const int nb = blockIdx.x * 4;

  // gather m_in = [h[dst] | h[src] | ea | 0-pad] into A-swizzled LDS tiles
  for (int idx = tid; idx < 48 * (KP / 8); idx += 128) {     // 1728 8-half chunks
    const int row = idx / (KP / 8);
    const int a8 = idx - row * (KP / 8);
    const int k0 = a8 * 8;
    const int dstn = nb + row / Kn;
    const int e = dstn * Kn + (row % Kn);
    float v[8];
    if (k0 < DM) {
#pragma unroll
      for (int j = 0; j < 8; ++j) v[j] = h[(size_t)dstn * DM + k0 + j];
    } else if (k0 < 2 * DM) {
      const int srcn = srcIdx[e];
#pragma unroll
      for (int j = 0; j < 8; ++j) v[j] = h[(size_t)srcn * DM + (k0 - DM) + j];
    } else if (k0 == 2 * DM) {
#pragma unroll
      for (int j = 0; j < 4; ++j) v[j] = ea[(size_t)e * 4 + j];
#pragma unroll
      for (int j = 4; j < 8; ++j) v[j] = 0.f;
    } else {
#pragma unroll
      for (int j = 0; j < 8; ++j) v[j] = 0.f;
    }
    const int rt = row >> 4, m = row & 15;
    const int st = k0 >> 5;
    const int ap = (k0 >> 3) & 3;
    _Float16* dp = s_in + ((rt * 9 + st) * 512) + (m + 16 * (ap & 1)) * 16 + (ap >> 1) * 8;
#pragma unroll
    for (int j = 0; j < 8; ++j) dp[j] = (_Float16)v[j];
  }
  __syncthreads();

  const int lane = tid & 31;
  const int fn = lane & 15, fm = lane >> 4;
  const v16h* B1 = (const v16h*)W1p;
  const v16h* B2 = (const v16h*)W2p;

  // stage 1: t = silu(m_in @ W1 + b1)  (48 x 256); 16 col tiles over 4 waves.
  // 3 row-tile accumulators live; B loaded once per k-step, reused 3x.
#pragma clang loop unroll(disable)
  for (int cc = 0; cc < 4; ++cc) {
    const int ct = wave * 4 + cc;
    v8f ac0 = {}, ac1 = {}, ac2 = {};
#pragma unroll
    for (int s = 0; s < KP / 32; ++s) {
      const v16h b = B1[(s * (DH / 16) + ct) * 32 + lane];
      ac0 = WMMA_F16(lds_frag(s_in, (0 * 9 + s) * 512 + lane * 16), b, ac0);
      ac1 = WMMA_F16(lds_frag(s_in, (1 * 9 + s) * 512 + lane * 16), b, ac1);
      ac2 = WMMA_F16(lds_frag(s_in, (2 * 9 + s) * 512 + lane * 16), b, ac2);
    }
    const float bb = b1[ct * 16 + fn];
    // store into stage-2 A-swizzled layout
    _Float16* tp = s_t + ((ct >> 1) * 512) + (16 * (fn >> 3)) * 16 + (fn & 7) + 8 * (ct & 1);
#pragma unroll
    for (int r = 0; r < 8; ++r) {
      const int mo = (fm * 8 + r) * 16;
      tp[0 * 8 * 512 + mo] = (_Float16)silu_f(ac0[r] + bb);
      tp[1 * 8 * 512 + mo] = (_Float16)silu_f(ac1[r] + bb);
      tp[2 * 8 * 512 + mo] = (_Float16)silu_f(ac2[r] + bb);
    }
  }
  __syncthreads();

  // stage 2: m = t @ W2 + b2  (48 x 128); 8 col tiles over 4 waves
#pragma clang loop unroll(disable)
  for (int cc = 0; cc < 2; ++cc) {
    const int ct = wave * 2 + cc;
    v8f ac0 = {}, ac1 = {}, ac2 = {};
#pragma unroll
    for (int s = 0; s < DH / 32; ++s) {
      const v16h b = B2[(s * (DM / 16) + ct) * 32 + lane];
      ac0 = WMMA_F16(lds_frag(s_t, (0 * 8 + s) * 512 + lane * 16), b, ac0);
      ac1 = WMMA_F16(lds_frag(s_t, (1 * 8 + s) * 512 + lane * 16), b, ac1);
      ac2 = WMMA_F16(lds_frag(s_t, (2 * 8 + s) * 512 + lane * 16), b, ac2);
    }
    const int col = ct * 16 + fn;
    const float bb = b2[col];
#pragma unroll
    for (int r = 0; r < 8; ++r) {
      const int mo = fm * 8 + r;
      s_m[(0 * 16 + mo) * DM + col] = ac0[r] + bb;
      s_m[(1 * 16 + mo) * DM + col] = ac1[r] + bb;
      s_m[(2 * 16 + mo) * DM + col] = ac2[r] + bb;
    }
  }
  __syncthreads();

  // segment-sum: edges are dst-major -> each node sums its 12 rows, no atomics
  for (int o = tid; o < 4 * DM; o += 128) {
    const int i = o >> 7, c = o & (DM - 1);
    float sum = 0.f;
#pragma unroll
    for (int j = 0; j < Kn; ++j) sum += s_m[(i * Kn + j) * DM + c];
    agg[(size_t)(nb + i) * DM + c] = sum;
  }
}

// ---------------- update MLP + residual (WMMA): h += mlp2([h|agg]) ----------------
__global__ void upd_kernel(float* __restrict__ h, const float* __restrict__ agg,
                           const _Float16* __restrict__ W1p, const float* __restrict__ b1,
                           const _Float16* __restrict__ W2p, const float* __restrict__ b2) {
  __shared__ __align__(32) _Float16 s_in[8 * 512];
  __shared__ __align__(32) _Float16 s_t[8 * 512];
  const int tid = threadIdx.x;
  const int wave = tid >> 5;
  const int nb = blockIdx.x * 16;

  for (int idx = tid; idx < 16 * (DH / 8); idx += 128) {   // 512 chunks
    const int row = idx >> 5;
    const int k0 = (idx & 31) * 8;
    float v[8];
    if (k0 < DM) {
#pragma unroll
      for (int j = 0; j < 8; ++j) v[j] = h[(size_t)(nb + row) * DM + k0 + j];
    } else {
#pragma unroll
      for (int j = 0; j < 8; ++j) v[j] = agg[(size_t)(nb + row) * DM + (k0 - DM) + j];
    }
    const int st = k0 >> 5;
    const int ap = (k0 >> 3) & 3;
    _Float16* dp = s_in + (st * 512) + (row + 16 * (ap & 1)) * 16 + (ap >> 1) * 8;
#pragma unroll
    for (int j = 0; j < 8; ++j) dp[j] = (_Float16)v[j];
  }
  __syncthreads();

  const int lane = tid & 31;
  const int fn = lane & 15, fm = lane >> 4;
  const v16h* B1 = (const v16h*)W1p;
  const v16h* B2 = (const v16h*)W2p;

#pragma clang loop unroll(disable)
  for (int cc = 0; cc < 4; ++cc) {          // stage 1: 16x256 hidden
    const int ct = wave * 4 + cc;
    v8f acc = {};
#pragma unroll
    for (int s = 0; s < DH / 32; ++s) {
      const v16h b = B1[(s * (DH / 16) + ct) * 32 + lane];
      acc = WMMA_F16(lds_frag(s_in, s * 512 + lane * 16), b, acc);
    }
    const float bb = b1[ct * 16 + fn];
    _Float16* tp = s_t + ((ct >> 1) * 512) + (16 * (fn >> 3)) * 16 + (fn & 7) + 8 * (ct & 1);
#pragma unroll
    for (int r = 0; r < 8; ++r)
      tp[(fm * 8 + r) * 16] = (_Float16)silu_f(acc[r] + bb);
  }
  __syncthreads();

#pragma clang loop unroll(disable)
  for (int cc = 0; cc < 2; ++cc) {          // stage 2: 16x128, residual add
    const int ct = wave * 2 + cc;
    v8f acc = {};
#pragma unroll
    for (int s = 0; s < DH / 32; ++s) {
      const v16h b = B2[(s * (DM / 16) + ct) * 32 + lane];
      acc = WMMA_F16(lds_frag(s_t, s * 512 + lane * 16), b, acc);
    }
    const int col = ct * 16 + fn;
    const float bb = b2[col];
#pragma unroll
    for (int r = 0; r < 8; ++r) {
      const size_t off = (size_t)(nb + fm * 8 + r) * DM + col;
      h[off] = h[off] + acc[r] + bb;
    }
  }
}

// ---------------- graph norm ----------------
__global__ void gnorm_kernel(float* __restrict__ h, const float* __restrict__ w,
                             const float* __restrict__ b, const float* __restrict__ ms) {
  const int g = blockIdx.x, c = threadIdx.x;
  float sum = 0.f;
  for (int n = 0; n < NPG; ++n) sum += h[(size_t)(g * NPG + n) * DM + c];
  const float msub = (sum / (float)NPG) * ms[c];
  float var = 0.f;
  for (int n = 0; n < NPG; ++n) {
    const size_t off = (size_t)(g * NPG + n) * DM + c;
    const float o = h[off] - msub;
    h[off] = o;
    var += o * o;
  }
  var /= (float)NPG;
  const float sc = w[c] * rsqrtf(var + 1e-5f);
  const float bb = b[c];
  for (int n = 0; n < NPG; ++n) {
    const size_t off = (size_t)(g * NPG + n) * DM + c;
    h[off] = sc * h[off] + bb;
  }
}

// ---------------- gate logit per node ----------------
__global__ void gate_kernel(const float* __restrict__ h,
                            const float* __restrict__ W1, const float* __restrict__ b1,
                            const float* __restrict__ W2, const float* __restrict__ b2,
                            float* __restrict__ gl) {
  __shared__ float s_h[DM], s_r[DM];
  const int node = blockIdx.x, j = threadIdx.x;
  s_h[j] = h[(size_t)node * DM + j];
  __syncthreads();
  float acc = b1[j];
#pragma unroll 8
  for (int k = 0; k < DM; ++k) acc += s_h[k] * W1[(size_t)k * DM + j];
  s_r[j] = tanhf(acc) * W2[j];
  __syncthreads();
  for (int st = 64; st > 0; st >>= 1) {
    if (j < st) s_r[j] += s_r[j + st];
    __syncthreads();
  }
  if (j == 0) gl[node] = s_r[0] + b2[0];
}

// ---------------- segment softmax + attention pool ----------------
__global__ void pool_kernel(const float* __restrict__ h, const float* __restrict__ gl,
                            float* __restrict__ pooled) {
  __shared__ float s_e[NPG];
  __shared__ float s_r[256];
  const int g = blockIdx.x, tid = threadIdx.x;
  const float* glg = gl + (size_t)g * NPG;
  float lmax = -3.0e38f;
  for (int n = tid; n < NPG; n += 256) lmax = fmaxf(lmax, glg[n]);
  s_r[tid] = lmax; __syncthreads();
  for (int st = 128; st > 0; st >>= 1) {
    if (tid < st) s_r[tid] = fmaxf(s_r[tid], s_r[tid + st]);
    __syncthreads();
  }
  const float gmax = s_r[0];
  __syncthreads();
  float lsum = 0.f;
  for (int n = tid; n < NPG; n += 256) {
    const float e = __expf(glg[n] - gmax);
    s_e[n] = e; lsum += e;
  }
  s_r[tid] = lsum; __syncthreads();
  for (int st = 128; st > 0; st >>= 1) {
    if (tid < st) s_r[tid] += s_r[tid + st];
    __syncthreads();
  }
  const float inv = 1.f / s_r[0];
  __syncthreads();
  if (tid < DM) {
    float acc = 0.f;
    for (int n = 0; n < NPG; ++n) acc += s_e[n] * h[(size_t)(g * NPG + n) * DM + tid];
    pooled[(size_t)g * DM + tid] = acc * inv;
  }
}

// ---------------- head MLP per graph ----------------
__global__ void head_kernel(const float* __restrict__ pooled,
                            const float* __restrict__ W1, const float* __restrict__ b1,
                            const float* __restrict__ W2, const float* __restrict__ b2,
                            float* __restrict__ out) {
  __shared__ float s_p[DM], s_r[DM];
  const int g = blockIdx.x, j = threadIdx.x;
  s_p[j] = pooled[(size_t)g * DM + j];
  __syncthreads();
  float acc = b1[j];
#pragma unroll 8
  for (int k = 0; k < DM; ++k) acc += s_p[k] * W1[(size_t)k * DM + j];
  s_r[j] = silu_f(acc) * W2[j];
  __syncthreads();
  for (int st = 64; st > 0; st >>= 1) {
    if (j < st) s_r[j] += s_r[j + st];
    __syncthreads();
  }
  if (j == 0) out[g] = s_r[0] + b2[0];
}

extern "C" void kernel_launch(void* const* d_in, const int* in_sizes, int n_in,
                              void* d_out, int out_size, void* d_ws, size_t ws_size,
                              hipStream_t stream) {
  (void)in_sizes; (void)n_in; (void)out_size; (void)ws_size;
  const float* x       = (const float*)d_in[0];
  const float* in_W1   = (const float*)d_in[2];
  const float* in_b1   = (const float*)d_in[3];
  const float* in_W2   = (const float*)d_in[4];
  const float* in_b2   = (const float*)d_in[5];
  const float* msg_W1  = (const float*)d_in[6];
  const float* msg_b1  = (const float*)d_in[7];
  const float* msg_W2  = (const float*)d_in[8];
  const float* msg_b2  = (const float*)d_in[9];
  const float* upd_W1  = (const float*)d_in[10];
  const float* upd_b1  = (const float*)d_in[11];
  const float* upd_W2  = (const float*)d_in[12];
  const float* upd_b2  = (const float*)d_in[13];
  const float* norm_w  = (const float*)d_in[14];
  const float* norm_b  = (const float*)d_in[15];
  const float* norm_ms = (const float*)d_in[16];
  const float* gate_W1 = (const float*)d_in[17];
  const float* gate_b1 = (const float*)d_in[18];
  const float* gate_W2 = (const float*)d_in[19];
  const float* gate_b2 = (const float*)d_in[20];
  const float* head_W1 = (const float*)d_in[21];
  const float* head_b1 = (const float*)d_in[22];
  const float* head_W2 = (const float*)d_in[23];
  const float* head_b2 = (const float*)d_in[24];
  float* out = (float*)d_out;

  float* h      = (float*)d_ws;                     // N*DM
  float* agg    = h + (size_t)Nn * DM;              // N*DM
  float* ea     = agg + (size_t)Nn * DM;            // E*4
  float* gl     = ea + (size_t)En * 4;              // N
  float* pooled = gl + Nn;                          // B*DM
  int*   srcIdx = (int*)(pooled + (size_t)Bg * DM); // E ints
  _Float16* wp  = (_Float16*)(srcIdx + En);         // 3*LSTRIDE halves, 32B aligned

  for (int l = 0; l < 3; ++l) {
    _Float16* wl = wp + (size_t)l * LSTRIDE;
    pack_b_kernel<<<T_MSG1, 32, 0, stream>>>(msg_W1 + (size_t)l * DI * DH, wl,            DI, DH, DH / 16);
    pack_b_kernel<<<T_MSG2, 32, 0, stream>>>(msg_W2 + (size_t)l * DH * DM, wl + OFF_MSG2, DH, DM, DM / 16);
    pack_b_kernel<<<T_UPD1, 32, 0, stream>>>(upd_W1 + (size_t)l * 2 * DM * DH, wl + OFF_UPD1, DH, DH, DH / 16);
    pack_b_kernel<<<T_UPD2, 32, 0, stream>>>(upd_W2 + (size_t)l * DH * DM, wl + OFF_UPD2, DH, DM, DM / 16);
  }

  knn_kernel<<<Bg * 4, 256, 0, stream>>>(x, srcIdx, ea);
  input_mlp_kernel<<<Nn, DM, 0, stream>>>(x, in_W1, in_b1, in_W2, in_b2, h);
  for (int l = 0; l < 3; ++l) {
    const _Float16* wl = wp + (size_t)l * LSTRIDE;
    msg_kernel<<<Nn / 4, 128, 0, stream>>>(h, ea, srcIdx,
        wl, msg_b1 + (size_t)l * DH,
        wl + OFF_MSG2, msg_b2 + (size_t)l * DM, agg);
    upd_kernel<<<Nn / 16, 128, 0, stream>>>(h, agg,
        wl + OFF_UPD1, upd_b1 + (size_t)l * DH,
        wl + OFF_UPD2, upd_b2 + (size_t)l * DM);
    gnorm_kernel<<<Bg, DM, 0, stream>>>(h, norm_w + l * DM, norm_b + l * DM, norm_ms + l * DM);
  }
  gate_kernel<<<Nn, DM, 0, stream>>>(h, gate_W1, gate_b1, gate_W2, gate_b2, gl);
  pool_kernel<<<Bg, 256, 0, stream>>>(h, gl, pooled);
  head_kernel<<<Bg, DM, 0, stream>>>(pooled, head_W1, head_b1, head_W2, head_b2, out);
}
